// NerfRender_56435870269582
// MI455X (gfx1250) — matrix operand order repaired and checked by reading, not compiled, not run on hardware
//
#include <hip/hip_runtime.h>
#include <math.h>

typedef __attribute__((ext_vector_type(16))) _Float16 v16h;
typedef __attribute__((ext_vector_type(8)))  _Float16 v8h;
typedef __attribute__((ext_vector_type(8)))  float    v8f;

// ---------------- strides (f16 elems) ----------------
#define SA    296          // activation row stride, 288 cols + pad (148 DW -> bank-tiled)
#define SK0   72           // W0^T row stride (K=63 -> pad 64 -> 72)
#define SK1   264          // W1/W2^T row stride (K=256 -> 264)
#define SKC   296          // Wc1^T row stride (K=283 -> pad 288 -> 296)

// ---------------- d_ws layout (f16 elems): pre-transposed weights ----------------
#define WSO_W0T   0                        // 256 x SK0
#define WSO_W1T   (WSO_W0T + 256 * SK0)    // 256 x SK1
#define WSO_W2T   (WSO_W1T + 256 * SK1)    // 256 x SK1
#define WSO_WC1T  (WSO_W2T + 256 * SK1)    // 128 x SKC
#define WSO_TOTAL (WSO_WC1T + 128 * SKC)   // 191488 elems = 382976 B

// ---------------- LDS layout (bytes) ----------------
// Weights staged in N-halves -> half-size weight buffer -> ~154 KB/WG
// -> TWO workgroups per WGP (16 waves) instead of one.
#define SZ_ACT     (128 * SA * 2)            // 75776 B (single in-place act buffer)
#define OFF_ACT    0
#define OFF_WBUF   (OFF_ACT + SZ_ACT)
#define SZ_WHALF   (128 * SK1 * 2)           // 67584 B (largest half: W1/W2)
#define OFF_F32    (OFF_WBUF + SZ_WHALF)     // 143360 B

// f32 region sub-offsets (in floats)
#define FO_B0    0
#define FO_B1    256
#define FO_B2    512
#define FO_WS    768
#define FO_BC1   1024
#define FO_WC2   1152      // 384, transposed [3][128]
#define FO_TV    1536      // 129
#define FO_TS    1665      // 129
#define FO_DT    1794      // 128
#define FO_DE    1922      // 27
#define FO_SIG   1949      // 128
#define FO_COL   2077      // 384
#define FO_OD    2461      // 6
#define FO_SC    2467      // 128 (scan buffer)
#define F32_CNT  2595

#define SMEM_BYTES (OFF_F32 + F32_CNT * 4)   // 153740 B -> 2 WGs fit in 320 KB LDS

__device__ __forceinline__ v16h cat8(v8h a, v8h b) {
    return __builtin_shufflevector(a, b, 0,1,2,3,4,5,6,7,8,9,10,11,12,13,14,15);
}

// Async-copy a contiguous f16 region from global into LDS via CDNA5
// GLOBAL_LOAD_ASYNC_TO_LDS_B128 (GVS mode; VDST VGPR = LDS byte offset;
// tracked by ASYNCcnt). Dynamic LDS starts at offset 0 (no static LDS here).
__device__ __forceinline__ void stage_async(const _Float16* __restrict__ g,
                                            int ldsByteOff, int nbytes, int tid) {
    unsigned long long base = (unsigned long long)(uintptr_t)g;
    for (int off = tid * 16; off < nbytes; off += 256 * 16) {
        int l = ldsByteOff + off;
        asm volatile("global_load_async_to_lds_b128 %0, %1, %2"
                     :: "v"(l), "v"(off), "s"(base)
                     : "memory");
    }
    asm volatile("s_wait_asynccnt 0x0" ::: "memory");
}

// Load this wave's A-fragments (16 rows x 32K per tile) for NK k-tiles.
template<int NK>
__device__ __forceinline__ void load_afrags(const _Float16* __restrict__ actIn,
                                            v16h* afrag, int tid) {
    const int wave = tid >> 5;
    const int lane = tid & 31;
    const int hi16 = lane >> 4;
    const int arow = wave * 16 + (lane & 15); // A: lane holds row M = lane%16
#pragma unroll
    for (int kt = 0; kt < NK; ++kt) {
        // A 16x32 f16: lo lanes K {0..7,16..23}, hi lanes K {8..15,24..31}
        const _Float16* p = actIn + arow * SA + kt * 32 + hi16 * 8;
        v8h c0 = *(const v8h*)(p);
        v8h c1 = *(const v8h*)(p + 16);
        afrag[kt] = cat8(c0, c1);
    }
}

// Compute NTILES output n-tiles (one staged N-half) against pre-loaded A-frags.
// In-place safe: rows are wave-private and A-frags already live in VGPRs.
// n-tiles in PAIRS (two accumulator chains) to fill WMMA hazard slots.
template<int NK, int NTILES, int SKW>
__device__ __forceinline__ void mlp_half(const _Float16* __restrict__ wTh,
                                         const float*    __restrict__ bias,
                                         _Float16*       __restrict__ actOut,
                                         const v16h* afrag, int colBase, int tid) {
    const int wave = tid >> 5;
    const int lane = tid & 31;
    const int hi16 = lane >> 4;
    const int l16  = lane & 15;
#pragma unroll
    for (int nt = 0; nt < NTILES; nt += 2) {
        const int lc0 = nt * 16 + l16;   // local column within this half
        const int lc1 = lc0 + 16;
        v8f acc0 = {0.f,0.f,0.f,0.f,0.f,0.f,0.f,0.f};
        v8f acc1 = {0.f,0.f,0.f,0.f,0.f,0.f,0.f,0.f};
#pragma unroll
        for (int kt = 0; kt < NK; ++kt) {
            // B 32x16 f16: lane = column, lo lanes K {0..7,16..23}, hi lanes K {8..15,24..31}
            const _Float16* p0 = wTh + lc0 * SKW + kt * 32 + hi16 * 8;
            const _Float16* p1 = wTh + lc1 * SKW + kt * 32 + hi16 * 8;
            v16h bf0 = cat8(*(const v8h*)(p0), *(const v8h*)(p0 + 16));
            v16h bf1 = cat8(*(const v8h*)(p1), *(const v8h*)(p1 + 16));
            acc0 = __builtin_amdgcn_wmma_f32_16x16x32_f16(
                false, afrag[kt], false, bf0, (short)0, acc0, false, false);
            acc1 = __builtin_amdgcn_wmma_f32_16x16x32_f16(
                false, afrag[kt], false, bf1, (short)0, acc1, false, false);
        }
        const float bv0 = bias[colBase + lc0];
        const float bv1 = bias[colBase + lc1];
        const int row0  = wave * 16 + hi16 * 8; // C/D: VGPR v -> row
#pragma unroll
        for (int v = 0; v < 8; ++v) {
            float x0 = acc0[v] + bv0; x0 = x0 > 0.f ? x0 : 0.f;
            float x1 = acc1[v] + bv1; x1 = x1 > 0.f ? x1 : 0.f;
            actOut[(row0 + v) * SA + colBase + lc0] = (_Float16)x0;
            actOut[(row0 + v) * SA + colBase + lc1] = (_Float16)x1;
        }
    }
}

// One-time weight prep: transpose + f32->f16 into d_ws, in LDS-ready [N][SK] layout.
extern "C" __global__ void __launch_bounds__(256)
prep_weights(const float* __restrict__ W0, const float* __restrict__ W1,
             const float* __restrict__ W2, const float* __restrict__ Wc1,
             _Float16* __restrict__ ws) {
    int idx = blockIdx.x * 256 + threadIdx.x;
    if (idx < 256 * SK0) {
        int n = idx / SK0, k = idx - n * SK0;
        ws[WSO_W0T + idx] = (_Float16)((k < 63) ? W0[k * 256 + n] : 0.f);
    } else if (idx < WSO_W2T) {
        int j = idx - WSO_W1T;
        int n = j / SK1, k = j - n * SK1;
        ws[idx] = (_Float16)((k < 256) ? W1[k * 256 + n] : 0.f);
    } else if (idx < WSO_WC1T) {
        int j = idx - WSO_W2T;
        int n = j / SK1, k = j - n * SK1;
        ws[idx] = (_Float16)((k < 256) ? W2[k * 256 + n] : 0.f);
    } else if (idx < WSO_TOTAL) {
        int j = idx - WSO_WC1T;
        int n = j / SKC, k = j - n * SKC;
        ws[idx] = (_Float16)((k < 283) ? Wc1[k * 128 + n] : 0.f);
    }
}

extern "C" __global__ void __launch_bounds__(256)
nerf_fused(const float* __restrict__ ray, const float* __restrict__ t,
           const _Float16* __restrict__ wsW,
           const float* __restrict__ b0, const float* __restrict__ b1,
           const float* __restrict__ b2,
           const float* __restrict__ Ws, const float* __restrict__ bs,
           const float* __restrict__ bc1,
           const float* __restrict__ Wc2, const float* __restrict__ bc2,
           float* __restrict__ outRgb, float* __restrict__ outWi) {
    extern __shared__ char smem[];
    _Float16* act  = (_Float16*)(smem + OFF_ACT);
    _Float16* wbuf = (_Float16*)(smem + OFF_WBUF);
    float* f32s = (float*)(smem + OFF_F32);
    float* b0L  = f32s + FO_B0;  float* b1L  = f32s + FO_B1;
    float* b2L  = f32s + FO_B2;  float* WsL  = f32s + FO_WS;
    float* bc1L = f32s + FO_BC1; float* Wc2L = f32s + FO_WC2;
    float* tvL  = f32s + FO_TV;  float* tsL  = f32s + FO_TS;
    float* dtL  = f32s + FO_DT;  float* deL  = f32s + FO_DE;
    float* sigL = f32s + FO_SIG; float* colL = f32s + FO_COL;
    float* odL  = f32s + FO_OD;  float* scL  = f32s + FO_SC;

    const int r   = blockIdx.x;   // one ray per block (128 samples = M-tile)
    const int tid = threadIdx.x;
    v16h af[9];                   // A-fragments for current layer (max 9 k-tiles)

    // ---- P0: stage scalars/biases; Wc2 transposed to [3][128]; async W0^T half0 ----
    for (int i = tid; i < 129; i += 256) tvL[i] = t[r * 129 + i];
    if (tid < 6) odL[tid] = ray[r * 6 + tid];
    b0L[tid] = b0[tid]; b1L[tid] = b1[tid]; b2L[tid] = b2[tid]; WsL[tid] = Ws[tid];
    if (tid < 128) bc1L[tid] = bc1[tid];
    for (int i = tid; i < 384; i += 256) {
        int ch = i >> 7, c = i & 127;
        Wc2L[ch * 128 + c] = Wc2[c * 3 + ch];
    }
    stage_async(wsW + WSO_W0T, OFF_WBUF, 128 * SK0 * 2, tid);
    __syncthreads();

    // ---- P1: rank-sort t (129 vals), dir posenc, point posenc ----
    if (tid < 129) {
        float v = tvL[tid];
        int rank = 0;
        for (int j = 0; j < 129; ++j) {
            float u = tvL[j];
            rank += (u < v) || (u == v && j < tid);
        }
        tsL[rank] = v;
    }
    if (tid < 27) {  // de = posenc(d, L=4)
        float v;
        if (tid < 3) v = odL[3 + tid];
        else {
            int g = tid - 3, l = g / 6, m = g % 6, dim = m % 3;
            float a = odL[3 + dim] * (float)(1 << l);
            v = (m < 3) ? __sinf(a) : __cosf(a);
        }
        deL[tid] = v;
    }
    for (int idx = tid; idx < 128 * 64; idx += 256) {  // xe: 63 feats, pad to 64
        int s = idx >> 6, f = idx & 63;
        float val = 0.f;
        if (f < 63) {
            float tt = tvL[s + 1];                  // x uses UNSORTED t[:,1:]
            if (f < 3) val = odL[f] + tt * odL[3 + f];
            else {
                int g = f - 3, l = g / 6, m = g % 6, dim = m % 3;
                float x = odL[dim] + tt * odL[3 + dim];
                float a = x * (float)(1 << l);
                val = (m < 3) ? __sinf(a) : __cosf(a);
            }
        }
        act[s * SA + f] = (_Float16)val;
    }
    __syncthreads();

    // ---- Layer 0 (63->256), in place, N staged in two 128-col halves ----
    if (tid < 128) dtL[tid] = tsL[tid + 1] - tsL[tid];
    load_afrags<2>(act, af, tid);
    mlp_half<2, 8, SK0>(wbuf, b0L, act, af, 0, tid);
    __syncthreads();
    stage_async(wsW + WSO_W0T + 128 * SK0, OFF_WBUF, 128 * SK0 * 2, tid);
    __syncthreads();
    mlp_half<2, 8, SK0>(wbuf, b0L, act, af, 128, tid);
    __syncthreads();

    // ---- Layer 1 (256->256) ----
    load_afrags<8>(act, af, tid);      // own rows only: safe pre-barrier
    stage_async(wsW + WSO_W1T, OFF_WBUF, 128 * SK1 * 2, tid);
    __syncthreads();
    mlp_half<8, 8, SK1>(wbuf, b1L, act, af, 0, tid);
    __syncthreads();
    stage_async(wsW + WSO_W1T + 128 * SK1, OFF_WBUF, 128 * SK1 * 2, tid);
    __syncthreads();
    mlp_half<8, 8, SK1>(wbuf, b1L, act, af, 128, tid);
    __syncthreads();

    // ---- Layer 2 (256->256) ----
    load_afrags<8>(act, af, tid);
    stage_async(wsW + WSO_W2T, OFF_WBUF, 128 * SK1 * 2, tid);
    __syncthreads();
    mlp_half<8, 8, SK1>(wbuf, b2L, act, af, 0, tid);
    __syncthreads();
    stage_async(wsW + WSO_W2T + 128 * SK1, OFF_WBUF, 128 * SK1 * 2, tid);
    __syncthreads();
    mlp_half<8, 8, SK1>(wbuf, b2L, act, af, 128, tid);
    __syncthreads();

    // ---- sigma head (softplus, vectorized v8h); append de (cols 256..287) ----
    if (tid < 128) {
        float acc = bs[0];
        const _Float16* hp = act + tid * SA;
#pragma unroll 4
        for (int c = 0; c < 256; c += 8) {
            v8h hv = *(const v8h*)(hp + c);
#pragma unroll
            for (int j = 0; j < 8; ++j) acc += (float)hv[j] * WsL[c + j];
        }
        sigL[tid] = (acc > 20.f) ? acc : log1pf(__expf(acc));
    }
    for (int idx = tid; idx < 128 * 32; idx += 256) {
        int s = idx >> 5, c = idx & 31;
        act[s * SA + 256 + c] = (_Float16)((c < 27) ? deL[c] : 0.f);
    }
    stage_async(wsW + WSO_WC1T, OFF_WBUF, 64 * SKC * 2, tid);
    __syncthreads();

    // ---- Color layer (288->128), in place, two 64-col halves ----
    load_afrags<9>(act, af, tid);
    mlp_half<9, 4, SKC>(wbuf, bc1L, act, af, 0, tid);
    __syncthreads();
    stage_async(wsW + WSO_WC1T + 64 * SKC, OFF_WBUF, 64 * SKC * 2, tid);
    __syncthreads();
    mlp_half<9, 4, SKC>(wbuf, bc1L, act, af, 64, tid);
    __syncthreads();

    // ---- color head (128->3, vectorized) + sigmoid ----
    for (int idx = tid; idx < 384; idx += 256) {
        int s = idx & 127, ch = idx >> 7;
        float acc = bc2[ch];
        const _Float16* hp = act + s * SA;
        const float* wp = Wc2L + ch * 128;
#pragma unroll 4
        for (int c = 0; c < 128; c += 8) {
            v8h hv = *(const v8h*)(hp + c);
#pragma unroll
            for (int j = 0; j < 8; ++j) acc += (float)hv[j] * wp[c + j];
        }
        colL[s * 3 + ch] = 1.f / (1.f + __expf(-acc));
    }

    // ---- volume rendering: parallel inclusive scan of sigma*dt over 128 ----
    if (tid < 128) scL[tid] = sigL[tid] * dtL[tid];
    __syncthreads();
#pragma unroll
    for (int s = 1; s < 128; s <<= 1) {
        float add = (tid < 128 && tid >= s) ? scL[tid - s] : 0.f;
        __syncthreads();
        if (tid < 128) scL[tid] += add;
        __syncthreads();
    }
    if (tid < 128) {
        float sdt = sigL[tid] * dtL[tid];
        float Ti  = __expf(-(scL[tid] - sdt));     // exp(-exclusive prefix)
        float w   = Ti * (1.f - __expf(-sdt));
        outWi[r * 128 + tid] = w;
        sigL[tid] = w;                              // reuse as weight buffer
    }
    __syncthreads();
    if (tid < 3) {                                  // per-channel reduction
        float acc = 0.f;
        for (int i = 0; i < 128; ++i) acc += sigL[i] * colL[i * 3 + tid];
        outRgb[r * 3 + tid] = acc;
    }
}

extern "C" void kernel_launch(void* const* d_in, const int* in_sizes, int n_in,
                              void* d_out, int out_size, void* d_ws, size_t ws_size,
                              hipStream_t stream) {
    const float* ray = (const float*)d_in[0];
    const float* t   = (const float*)d_in[1];
    const float* W0  = (const float*)d_in[2];
    const float* b0  = (const float*)d_in[3];
    const float* W1  = (const float*)d_in[4];
    const float* b1  = (const float*)d_in[5];
    const float* W2  = (const float*)d_in[6];
    const float* b2  = (const float*)d_in[7];
    const float* Ws  = (const float*)d_in[8];
    const float* bs  = (const float*)d_in[9];
    const float* Wc1 = (const float*)d_in[10];
    const float* bc1 = (const float*)d_in[11];
    const float* Wc2 = (const float*)d_in[12];
    const float* bc2 = (const float*)d_in[13];

    const int R = in_sizes[0] / 6;            // 4096 rays
    float* outRgb = (float*)d_out;            // [R,3]
    float* outWi  = (float*)d_out + R * 3;    // [R,128]
    _Float16* ws  = (_Float16*)d_ws;          // 382976 B pre-transposed f16 weights

    prep_weights<<<(WSO_TOTAL + 255) / 256, 256, 0, stream>>>(W0, W1, W2, Wc1, ws);
    nerf_fused<<<R, 256, SMEM_BYTES, stream>>>(ray, t, ws, b0, b1, b2,
                                               Ws, bs, bc1, Wc2, bc2,
                                               outRgb, outWi);
}
